// MultiheadAttention_85255100826237
// MI455X (gfx1250) — matrix operand order repaired and checked by reading, not compiled, not run on hardware
//
#include <hip/hip_runtime.h>

typedef __attribute__((ext_vector_type(16))) _Float16     v16h;
typedef __attribute__((ext_vector_type(8)))  _Float16     v8h;
typedef __attribute__((ext_vector_type(8)))  float        v8f;
typedef __attribute__((ext_vector_type(4)))  unsigned int v4u;
typedef __attribute__((ext_vector_type(4)))  int          v4i;
typedef __attribute__((ext_vector_type(8)))  int          v8i;

#define B_  4
#define S_  2048
#define D_  1024
#define H_  16
#define HD_ 64

#if defined(__has_builtin)
# if __has_builtin(__builtin_amdgcn_tensor_load_to_lds)
#  define HAVE_TDM 1
# endif
#endif
#ifndef HAVE_TDM
# define HAVE_TDM 0
#endif

// A-matrix (16x32 f16) per-lane K mapping: lane = (hi,lo), element e.
// VGPR0..3 hold K = hi*8 + 0..7 ; VGPR4..7 hold K = 16 + hi*8 + 0..7
__device__ __forceinline__ int amap(int e, int hi) {
  return (e < 8) ? (hi * 8 + e) : (16 + hi * 8 + (e - 8));
}

__device__ __forceinline__ v8f wmma_f16(v16h a, v16h b, v8f c) {
  return __builtin_amdgcn_wmma_f32_16x16x32_f16(
      false, a, false, b, (short)0, c, false, false);
}

// ---------------------------------------------------------------------------
// K1: q/k/v = x @ W^T + b  (NT GEMM, f16 into WMMA, f32 accumulate)
// q,k layout: [B,H,S,HD] f16.  v layout: TRANSPOSED [B,H,HD,S] f16 so the
// attention P@V B-operand loads are K-contiguous.
// A tile (x rows) is shared by all 4 waves of the WG -> staged into LDS by
// the Tensor Data Mover (2D descriptor, 32x16 f32 tile), s_wait_tensorcnt.
// grid: (M/16=512, Ntiles/4=16, 3), block: 128 (4 waves, one N-tile each)
// ---------------------------------------------------------------------------
__global__ __launch_bounds__(128) void qkv_proj_kernel(
    const float* __restrict__ x,
    const float* __restrict__ Wq, const float* __restrict__ bq,
    const float* __restrict__ Wk, const float* __restrict__ bk,
    const float* __restrict__ Wv, const float* __restrict__ bv,
    _Float16* __restrict__ qo, _Float16* __restrict__ ko, _Float16* __restrict__ vo)
{
#if HAVE_TDM
  __shared__ float xtile[16 * 32];
#endif
  const int lane = threadIdx.x & 31;
  const int wave = threadIdx.x >> 5;
  const int lo = lane & 15, hi = lane >> 4;
  const int mBase = blockIdx.x * 16;
  const int nBase = (blockIdx.y * 4 + wave) * 16;

  const float* W; const float* bias; _Float16* dst;
  if (blockIdx.z == 0)      { W = Wq; bias = bq; dst = qo; }
  else if (blockIdx.z == 1) { W = Wk; bias = bk; dst = ko; }
  else                      { W = Wv; bias = bv; dst = vo; }

  const float* xrow = x + (size_t)(mBase + lo) * D_;
  const float* wrow = W + (size_t)(nBase + lo) * D_;

  v8f c = {};
  for (int kb = 0; kb < D_; kb += 32) {
#if HAVE_TDM
    if (wave == 0) {
      // Tensor DMA: 16 rows x 32 cols f32 tile of x -> LDS (row-major)
      const unsigned ldsoff = (unsigned)(uintptr_t)&xtile[0];
      const unsigned long long ga =
          (unsigned long long)(uintptr_t)(x + (size_t)mBase * D_ + kb);
      v4u g0 = { 1u,                                   // count=1, user mode
                 ldsoff,                               // lds_addr
                 (unsigned)ga,                         // global_addr[31:0]
                 (unsigned)((ga >> 32) & 0x01FFFFFFu) | (2u << 30) }; // type=2
      v8i g1 = { (int)(2u << 16),      // wg_mask=0, data_size=2 (4 bytes)
                 (int)((unsigned)D_ << 16),        // tensor_dim0[15:0]=1024
                 (int)((unsigned)(B_ * S_) << 16), // tensor_dim1[15:0]=8192
                 (int)(32u << 16),                 // tile_dim0=32
                 16,                               // tile_dim1=16
                 D_,                               // tensor_dim0_stride=1024
                 0, 0 };
      v4i gz = { 0, 0, 0, 0 };
#if __clang_major__ >= 23
      v8i gz8 = { 0, 0, 0, 0, 0, 0, 0, 0 };
      __builtin_amdgcn_tensor_load_to_lds(g0, g1, gz, gz, gz8, 0);
#else
      __builtin_amdgcn_tensor_load_to_lds(g0, g1, gz, gz, 0);
#endif
      __builtin_amdgcn_s_wait_tensorcnt(0);
    }
    __syncthreads();
#endif
    v16h a, bm;
#pragma unroll
    for (int e = 0; e < 16; ++e) {
#if HAVE_TDM
      a[e]  = (_Float16)xtile[lo * 32 + amap(e, hi)];
#else
      a[e]  = (_Float16)xrow[kb + amap(e, hi)];
#endif
      bm[e] = (_Float16)wrow[kb + amap(e, hi)];
    }
    c = wmma_f16(a, bm, c);
#if HAVE_TDM
    __syncthreads();   // protect xtile before next TDM overwrite
#endif
  }
  (void)xrow;

  const int n  = nBase + lo;
  const float bb = bias[n];
  const int h = n >> 6, hd = n & 63;
#pragma unroll
  for (int r = 0; r < 8; ++r) {
    const int m = mBase + r + hi * 8;
    const int b = m >> 11, s = m & (S_ - 1);
    const _Float16 val = (_Float16)(c[r] + bb);
    if (blockIdx.z == 2)   // V transposed: [B,H,HD,S]
      dst[(((size_t)(b * H_ + h)) * HD_ + hd) * S_ + s] = val;
    else                   // Q,K: [B,H,S,HD]
      dst[(((size_t)(b * H_ + h)) * S_ + s) * HD_ + hd] = val;
  }
}

// ---------------------------------------------------------------------------
// K2 (fused): per wave = 16 q-rows x full 2048 k.
//  pass 1: scores = q@k^T (WMMA), online row (max,sumexp) in registers only.
//  pass 2: recompute scores, p = exp(s-m)/l, single NON-TEMPORAL probs write
//          (1.07 GB streamed past L2, keeping q/k/vt L2-resident), P routed
//          C-layout -> A-layout through 1KB LDS, ctx += P @ V^T (WMMA),
//          out written straight from accumulators (also NT).
// grid: (S/16/4=32, B*H=64), block 128 (4 independent waves).
// ---------------------------------------------------------------------------
__global__ __launch_bounds__(128) void attn_kernel(
    const _Float16* __restrict__ q, const _Float16* __restrict__ k,
    const _Float16* __restrict__ vt,
    float* __restrict__ probs, float* __restrict__ out)
{
  __shared__ _Float16 pbuf[4][16][32];
  const int lane = threadIdx.x & 31;
  const int wave = threadIdx.x >> 5;
  const int lo = lane & 15, hi = lane >> 4;
  const int bh = blockIdx.y;
  const int b = bh >> 4, h = bh & (H_ - 1);
  const int qBase = (blockIdx.x * 4 + wave) * 16;

  // A operand: q rows, reused across all k. K=64 -> two K=32 halves.
  const _Float16* qrow = q + ((size_t)bh * S_ + qBase + lo) * HD_;
  const v8h qa0 = *(const v8h*)(qrow + hi * 16);
  const v8h qa1 = *(const v8h*)(qrow + 8 + hi * 16);
  const v8h qa2 = *(const v8h*)(qrow + 32 + hi * 16);
  const v8h qa3 = *(const v8h*)(qrow + 40 + hi * 16);
  v16h a0, a1;
#pragma unroll
  for (int e = 0; e < 8; ++e) {
    a0[e] = qa0[e]; a0[e + 8] = qa1[e];
    a1[e] = qa2[e]; a1[e + 8] = qa3[e];
  }

  // ---- pass 1: online softmax stats, no global writes ----
  float m[8], l[8];
#pragma unroll
  for (int r = 0; r < 8; ++r) { m[r] = -__builtin_inff(); l[r] = 0.0f; }

  for (int kt = 0; kt < S_ / 16; ++kt) {
    const _Float16* krow = k + ((size_t)bh * S_ + kt * 16 + lo) * HD_;
    const v16h b0 = *(const v16h*)(krow + hi * 16);
    const v16h b1 = *(const v16h*)(krow + 32 + hi * 16);
    v8f c = {};
    c = wmma_f16(a0, b0, c);
    c = wmma_f16(a1, b1, c);
#pragma unroll
    for (int r = 0; r < 8; ++r) {
      const float xv = c[r];
      if (xv > m[r]) { l[r] = l[r] * __expf(m[r] - xv) + 1.0f; m[r] = xv; }
      else           { l[r] += __expf(xv - m[r]); }
    }
  }
  // merge partials across the 16 lanes (different k-columns) sharing each row
#pragma unroll
  for (int r = 0; r < 8; ++r) {
    float mm = m[r], ss = l[r];
#pragma unroll
    for (int off = 1; off < 16; off <<= 1) {
      const float mo = __shfl_xor(mm, off, 32);
      const float so = __shfl_xor(ss, off, 32);
      const float M  = fmaxf(mm, mo);
      ss = ss * __expf(mm - M) + so * __expf(mo - M);
      mm = M;
    }
    m[r] = mm;
    l[r] = 1.0f / ss;   // l[] now holds 1/sum
  }

  // ---- pass 2: recompute, normalize, write probs once (NT), ctx WMMA ----
  v8f acc[4] = {{}, {}, {}, {}};
  for (int kp = 0; kp < S_ / 32; ++kp) {
    const int kbase = kp * 32;
#pragma unroll
    for (int t = 0; t < 2; ++t) {
      const _Float16* krow = k + ((size_t)bh * S_ + kbase + t * 16 + lo) * HD_;
      const v16h b0 = *(const v16h*)(krow + hi * 16);
      const v16h b1 = *(const v16h*)(krow + 32 + hi * 16);
      v8f c = {};
      c = wmma_f16(a0, b0, c);
      c = wmma_f16(a1, b1, c);
#pragma unroll
      for (int r = 0; r < 8; ++r) {            // C: row = r+hi*8, col = lo
        const int row = r + hi * 8;
        const float p = __expf(c[r] - m[r]) * l[r];
        // streaming store: write-once, never re-read -> bypass L2 residency
        __builtin_nontemporal_store(
            p, &probs[((size_t)bh * S_ + qBase + row) * S_ + kbase + t * 16 + lo]);
        pbuf[wave][row][t * 16 + lo] = (_Float16)p;
      }
    }
    // P tile: C-layout -> A-layout via per-wave LDS (same-wave DS ordering)
    v16h ap;
#pragma unroll
    for (int e = 0; e < 16; ++e)
      ap[e] = pbuf[wave][lo][hi * 16 + e];
#pragma unroll
    for (int nc = 0; nc < 4; ++nc) {           // B: N=lo (hd), K contiguous
      const v16h bv = *(const v16h*)(vt + ((size_t)bh * HD_ + nc * 16 + lo) * S_
                                     + kbase + hi * 16);
      acc[nc] = wmma_f16(ap, bv, acc[nc]);
    }
  }

  // out: wave owns the full row strip -> no cross-wave reduction needed
#pragma unroll
  for (int nc = 0; nc < 4; ++nc)
#pragma unroll
    for (int r = 0; r < 8; ++r)
      __builtin_nontemporal_store(
          acc[nc][r],
          &out[((size_t)(b * S_ + qBase + r + hi * 8)) * D_ + h * HD_
               + nc * 16 + lo]);
}

// ---------------------------------------------------------------------------
extern "C" void kernel_launch(void* const* d_in, const int* in_sizes, int n_in,
                              void* d_out, int out_size, void* d_ws, size_t ws_size,
                              hipStream_t stream) {
  const float* x  = (const float*)d_in[0];
  const float* Wq = (const float*)d_in[1];
  const float* bq = (const float*)d_in[2];
  const float* Wk = (const float*)d_in[3];
  const float* bk = (const float*)d_in[4];
  const float* Wv = (const float*)d_in[5];
  const float* bv = (const float*)d_in[6];

  float* out   = (float*)d_out;                  // [B,S,D]
  float* probs = out + (size_t)B_ * S_ * D_;     // [B,H,S,S]

  const size_t qkvElems = (size_t)B_ * S_ * D_;  // 8,388,608
  _Float16* qbuf  = (_Float16*)d_ws;
  _Float16* kbuf  = qbuf + qkvElems;
  _Float16* vtbuf = kbuf + qkvElems;             // transposed V

  dim3 blk(128);
  dim3 g1((B_ * S_) / 16, D_ / 16 / 4, 3);
  qkv_proj_kernel<<<g1, blk, 0, stream>>>(x, Wq, bq, Wk, bk, Wv, bv,
                                          qbuf, kbuf, vtbuf);

  dim3 g2(S_ / 16 / 4, B_ * H_);
  attn_kernel<<<g2, blk, 0, stream>>>(qbuf, kbuf, vtbuf, probs, out);
}
